// DepthwiseDeformableConv_42666205118544
// MI455X (gfx1250) — matrix operand order repaired and checked by reading, not compiled, not run on hardware
//
#include <hip/hip_runtime.h>
#include <hip/hip_bf16.h>
#include <math.h>

// ---------------------------------------------------------------------------
// DeformableDepthwiseConv block for MI455X (gfx1250, wave32, WMMA).
// Conv stages run on v_wmma_f32_16x16x32_f16; gather/GN/GELU stages are
// bandwidth-bound fp32 VALU. All launches are exact multiples -> EXEC is all
// ones at every WMMA (ISA requirement).
// ---------------------------------------------------------------------------

typedef _Float16 v16h __attribute__((ext_vector_type(16)));
typedef _Float16 v8h  __attribute__((ext_vector_type(8)));
typedef float    v8f  __attribute__((ext_vector_type(8)));

#define Bn 4
#define Cn 64
#define Hn 128
#define Wn 128
#define HW (Hn*Wn)            // 16384
#define NPIX (Bn*HW)          // 65536
#define NELEM (Bn*Cn*HW)      // 4194304
#define GROUPS 32
#define CPG (Cn/GROUPS)       // 2
#define GN_N (CPG*HW)         // 32768 elements per (b,group)

// Workspace layout (bytes). Total ~50.5 MB.
#define OFF_XH      ((size_t)0)                       // f16 NHWC x copy: 8 MB
#define OFF_TOFF    ((size_t)8  << 20)                // f16 NHWC t_off:  8 MB
#define OFF_TMASK   ((size_t)16 << 20)                // f16 NHWC t_mask: 8 MB
#define OFF_OFFB    ((size_t)24 << 20)                // f32 off[pix][18]: 4.5 MB
#define OFF_MASKB   (OFF_OFFB + (size_t)NPIX*18*4)    // f32 mask[pix][9]: 2.25 MB
#define OFF_OUTPRE  ((size_t)32 << 20)                // f32 pre-GN: 16 MB
#define OFF_STATS   ((size_t)48 << 20)                // f32 stats[4*32*2]
#define OFF_PK      (OFF_STATS + 4096)                // packed B frags: 70*32*16 f16

// WMMA fragment inventory:
//   frag 0..15 : conv1x1   frag = br*8 + nt*2 + kc         (br: 0=off,1=mask)
//   frag 16..51: conv3 off frag = 16 + (tap*2+kc)*2 + nt
//   frag 52..69: conv3 msk frag = 52 + tap*2 + kc
#define NFRAG 70

__device__ __forceinline__ v8h hzero() {
  v8h z;
#pragma unroll
  for (int i = 0; i < 8; ++i) z[i] = (_Float16)0.0f;
  return z;
}

union AFrag { v16h v; v8h h[2]; };

// ----------------------------- init stats ----------------------------------
__global__ void k_init_stats(float* stats) {
  if (threadIdx.x < Bn * GROUPS * 2) stats[threadIdx.x] = 0.0f;
}

// ------------------------- pack WMMA B fragments ---------------------------
// CDNA5 16-bit B layout (32x16): lane L holds column N=L&15; hi = L>=16;
// element j of the lane's v16h is K = hi*8 + j for j<8, K = 16 + hi*8 + (j-8)
// for j>=8 (matches the A fragment order).
__global__ void k_pack(const float* __restrict__ w_off1,
                       const float* __restrict__ w_mask1,
                       const float* __restrict__ w_off2,
                       const float* __restrict__ w_mask2,
                       _Float16* __restrict__ pk) {
  int gid = blockIdx.x * 64 + threadIdx.x;
  if (gid >= NFRAG * 32) return;
  int frag = gid >> 5, lane = gid & 31;
  int N = lane & 15;
  int kbase = (lane >> 4) * 8;
  _Float16* dst = pk + (size_t)gid * 16;

  if (frag < 16) {                       // 1x1 convs: w1[nout*64 + kin]
    int br = frag >> 3, r = frag & 7, nt = r >> 1, kc = r & 1;
    const float* w1 = br ? w_mask1 : w_off1;
    int n = nt * 16 + N;
#pragma unroll
    for (int j = 0; j < 16; ++j) {
      int K = kc * 32 + kbase + (j < 8 ? j : j + 8);
      dst[j] = (_Float16)w1[n * 64 + K];
    }
  } else if (frag < 52) {                // 3x3 off conv: OIHW (18,64,3,3)
    int r = frag - 16;
    int nt = r & 1, tkc = r >> 1, kc = tkc & 1, tap = tkc >> 1;
    int ky = tap / 3, kx = tap % 3;
    int n = nt * 16 + N;
#pragma unroll
    for (int j = 0; j < 16; ++j) {
      int c = kc * 32 + kbase + (j < 8 ? j : j + 8);
      float v = (n < 18) ? w_off2[((n * 64 + c) * 3 + ky) * 3 + kx] : 0.0f;
      dst[j] = (_Float16)v;
    }
  } else {                               // 3x3 mask conv: OIHW (9,64,3,3)
    int r = frag - 52;
    int kc = r & 1, tap = r >> 1;
    int ky = tap / 3, kx = tap % 3;
#pragma unroll
    for (int j = 0; j < 16; ++j) {
      int c = kc * 32 + kbase + (j < 8 ? j : j + 8);
      float v = (N < 9) ? w_mask2[((N * 64 + c) * 3 + ky) * 3 + kx] : 0.0f;
      dst[j] = (_Float16)v;
    }
  }
}

// --------------------- x: NCHW f32 -> NHWC f16 copy ------------------------
__global__ void k_tox(const float* __restrict__ x, _Float16* __restrict__ xh) {
  size_t id = (size_t)blockIdx.x * 256 + threadIdx.x;   // < NELEM
  int p = (int)(id >> 6);
  int c = (int)(id & 63);
  int b = p >> 14, hw = p & (HW - 1);
  xh[id] = (_Float16)x[(((size_t)b * Cn + c) << 14) + hw];
}

// ----------------------------- conv 1x1 (GEMM) -----------------------------
// One wave = 16-pixel M-tile. A: lane L holds row M=L&15; two contiguous
// 8xf16 runs at channel offsets {hi*8, hi*8+16} (hi = L>=16) per K-chunk.
__global__ void __launch_bounds__(256)
k_conv1(const _Float16* __restrict__ xh, const _Float16* __restrict__ pk,
        _Float16* __restrict__ t_off, _Float16* __restrict__ t_mask) {
  int row = blockIdx.x;                       // b*128 + h
  int wave = threadIdx.x >> 5, lane = threadIdx.x & 31;
  int M = lane & 15, hi = lane >> 4;
  size_t pix = (size_t)row * Wn + wave * 16 + M;
  const _Float16* ap = xh + pix * Cn + hi * 8;
  AFrag a0, a1;
  a0.h[0] = *(const v8h*)(ap);
  a0.h[1] = *(const v8h*)(ap + 16);
  a1.h[0] = *(const v8h*)(ap + 32);
  a1.h[1] = *(const v8h*)(ap + 48);
  size_t tile_pix = (size_t)row * Wn + wave * 16;

#pragma unroll
  for (int br = 0; br < 2; ++br) {
    _Float16* tout = br ? t_mask : t_off;
#pragma unroll
    for (int nt = 0; nt < 4; ++nt) {
      v8f acc = {0.f, 0.f, 0.f, 0.f, 0.f, 0.f, 0.f, 0.f};
      v16h b0 = *(const v16h*)(pk + ((size_t)((br * 8 + nt * 2 + 0) * 32 + lane)) * 16);
      v16h b1 = *(const v16h*)(pk + ((size_t)((br * 8 + nt * 2 + 1) * 32 + lane)) * 16);
      acc = __builtin_amdgcn_wmma_f32_16x16x32_f16(false, a0.v, false, b0,
                                                   (short)0, acc, false, false);
      acc = __builtin_amdgcn_wmma_f32_16x16x32_f16(false, a1.v, false, b1,
                                                   (short)0, acc, false, false);
      int n = nt * 16 + (lane & 15);
#pragma unroll
      for (int r2 = 0; r2 < 8; ++r2) {
        int Mo = r2 + hi * 8;                 // C/D layout: lane N, VGPR->M
        tout[(tile_pix + Mo) * Cn + n] = (_Float16)acc[r2];
      }
    }
  }
}

// ------------------------ conv 3x3 (im2col GEMM) ---------------------------
// K = tap*64 + c. Per tap, A is the 1x1-conv output at the shifted pixel;
// out-of-image taps contribute zero (pad=1).
__global__ void __launch_bounds__(256)
k_conv3(const _Float16* __restrict__ t_off, const _Float16* __restrict__ t_mask,
        const _Float16* __restrict__ pk,
        float* __restrict__ offb, float* __restrict__ maskb) {
  int row = blockIdx.x;
  int b = row >> 7, h = row & 127;
  int wave = threadIdx.x >> 5, lane = threadIdx.x & 31;
  int M = lane & 15, hi = lane >> 4;
  int w0 = wave * 16;
  int wpix = w0 + M;
  size_t base_pix = ((size_t)b * Hn + h) * Wn;

  v8f accO0 = {0.f, 0.f, 0.f, 0.f, 0.f, 0.f, 0.f, 0.f};
  v8f accO1 = {0.f, 0.f, 0.f, 0.f, 0.f, 0.f, 0.f, 0.f};
  v8f accM  = {0.f, 0.f, 0.f, 0.f, 0.f, 0.f, 0.f, 0.f};

#pragma unroll
  for (int tap = 0; tap < 9; ++tap) {
    int dy = tap / 3 - 1, dx = tap % 3 - 1;
    int hh = h + dy, ww = wpix + dx;
    bool valid = ((unsigned)hh < (unsigned)Hn) && ((unsigned)ww < (unsigned)Wn);
    size_t spix = ((size_t)b * Hn + hh) * Wn + ww;
    const _Float16* ao = t_off  + spix * Cn + hi * 8;
    const _Float16* am = t_mask + spix * Cn + hi * 8;
#pragma unroll
    for (int kc = 0; kc < 2; ++kc) {
      AFrag aO, aM;
      if (valid) {
        aO.h[0] = *(const v8h*)(ao + kc * 32);
        aO.h[1] = *(const v8h*)(ao + kc * 32 + 16);
        aM.h[0] = *(const v8h*)(am + kc * 32);
        aM.h[1] = *(const v8h*)(am + kc * 32 + 16);
      } else {
        aO.h[0] = hzero(); aO.h[1] = hzero();
        aM.h[0] = hzero(); aM.h[1] = hzero();
      }
      int fO = 16 + (tap * 2 + kc) * 2;
      v16h bO0 = *(const v16h*)(pk + ((size_t)((fO + 0) * 32 + lane)) * 16);
      v16h bO1 = *(const v16h*)(pk + ((size_t)((fO + 1) * 32 + lane)) * 16);
      int fM = 52 + tap * 2 + kc;
      v16h bM  = *(const v16h*)(pk + ((size_t)(fM * 32 + lane)) * 16);
      accO0 = __builtin_amdgcn_wmma_f32_16x16x32_f16(false, aO.v, false, bO0,
                                                     (short)0, accO0, false, false);
      accO1 = __builtin_amdgcn_wmma_f32_16x16x32_f16(false, aO.v, false, bO1,
                                                     (short)0, accO1, false, false);
      accM  = __builtin_amdgcn_wmma_f32_16x16x32_f16(false, aM.v, false, bM,
                                                     (short)0, accM, false, false);
    }
  }

  int Nn = lane & 15;
#pragma unroll
  for (int r2 = 0; r2 < 8; ++r2) {
    int Mo = r2 + hi * 8;
    size_t p = base_pix + w0 + Mo;
    if (Nn < 18) offb[p * 18 + Nn] = accO0[r2];
    int n1 = 16 + Nn;
    if (n1 < 18) offb[p * 18 + n1] = accO1[r2];
    if (Nn < 9)  maskb[p * 9 + Nn] = 1.0f / (1.0f + expf(-accM[r2]));
  }
}

// ------------------ deformable bilinear gather + reduce --------------------
__device__ __forceinline__ float fetch_xc(const float* __restrict__ xc, int y, int x) {
  if ((unsigned)y < (unsigned)Hn && (unsigned)x < (unsigned)Wn)
    return xc[y * Wn + x];
  return 0.0f;
}

__global__ void __launch_bounds__(256)
k_deform(const float* __restrict__ x, const float* __restrict__ offb,
         const float* __restrict__ maskb, const float* __restrict__ wgt,
         const float* __restrict__ bias, float* __restrict__ out_pre,
         float* __restrict__ stats) {
  int id = blockIdx.x * 256 + threadIdx.x;     // < NELEM; pixel-major
  int p = id & (NPIX - 1);
  int c = id >> 16;
  int b = p >> 14;
  int hw = p & (HW - 1);
  int h = hw >> 7, w = hw & 127;
  const float* offp = offb + (size_t)p * 18;
  const float* mskp = maskb + (size_t)p * 9;
  const float* xc = x + (((size_t)b * Cn + c) << 14);

  float acc = 0.0f;
#pragma unroll
  for (int k = 0; k < 9; ++k) {
    int ky = k / 3 - 1, kx = k % 3 - 1;
    float py = offp[2 * k]     + (float)(h + ky);
    float px = offp[2 * k + 1] + (float)(w + kx);
    float y0f = floorf(py), x0f = floorf(px);
    float wy = py - y0f, wx = px - x0f;
    int y0 = (int)y0f, x0 = (int)x0f;
    float v00 = fetch_xc(xc, y0,     x0);
    float v01 = fetch_xc(xc, y0,     x0 + 1);
    float v10 = fetch_xc(xc, y0 + 1, x0);
    float v11 = fetch_xc(xc, y0 + 1, x0 + 1);
    float v = v00 * (1.f - wy) * (1.f - wx) + v01 * (1.f - wy) * wx
            + v10 * wy * (1.f - wx)         + v11 * wy * wx;
    acc += mskp[k] * wgt[c * 9 + k] * v;
  }
  acc += bias[c];
  out_pre[(((size_t)b * Cn + c) << 14) + hw] = acc;

  // (b, c) are uniform across the block (256 | 16384); reduce then 1 atomic.
  __shared__ float ssum, ssq;
  if (threadIdx.x == 0) { ssum = 0.0f; ssq = 0.0f; }
  __syncthreads();
  atomicAdd(&ssum, acc);
  atomicAdd(&ssq, acc * acc);
  __syncthreads();
  if (threadIdx.x == 0) {
    int g = c >> 1;   // CPG == 2
    atomicAdd(&stats[(b * GROUPS + g) * 2 + 0], ssum);
    atomicAdd(&stats[(b * GROUPS + g) * 2 + 1], ssq);
  }
}

// ------------------------- GroupNorm + exact GELU --------------------------
__global__ void __launch_bounds__(256)
k_gn_gelu(const float* __restrict__ out_pre, const float* __restrict__ stats,
          const float* __restrict__ gamma, const float* __restrict__ beta,
          float* __restrict__ out) {
  size_t e = (size_t)blockIdx.x * 256 + threadIdx.x;   // NCHW linear
  int c = (int)((e >> 14) & 63);
  int b = (int)(e >> 20);
  int g = c >> 1;
  float s  = stats[(b * GROUPS + g) * 2 + 0];
  float sq = stats[(b * GROUPS + g) * 2 + 1];
  float mean = s / (float)GN_N;
  float var = sq / (float)GN_N - mean * mean;
  float v = (out_pre[e] - mean) * rsqrtf(var + 1e-5f);
  v = v * gamma[c] + beta[c];
  out[e] = 0.5f * v * (1.0f + erff(v * 0.70710678118654752f));
}

// ---------------------------------------------------------------------------
extern "C" void kernel_launch(void* const* d_in, const int* in_sizes, int n_in,
                              void* d_out, int out_size, void* d_ws, size_t ws_size,
                              hipStream_t stream) {
  const float* x       = (const float*)d_in[0];
  const float* w_off1  = (const float*)d_in[1];
  const float* w_off2  = (const float*)d_in[2];
  const float* w_mask1 = (const float*)d_in[3];
  const float* w_mask2 = (const float*)d_in[4];
  const float* weight  = (const float*)d_in[5];
  const float* bias    = (const float*)d_in[6];
  const float* gamma   = (const float*)d_in[7];
  const float* beta    = (const float*)d_in[8];

  char* ws = (char*)d_ws;
  _Float16* xh     = (_Float16*)(ws + OFF_XH);
  _Float16* t_off  = (_Float16*)(ws + OFF_TOFF);
  _Float16* t_mask = (_Float16*)(ws + OFF_TMASK);
  float*    offb   = (float*)(ws + OFF_OFFB);
  float*    maskb  = (float*)(ws + OFF_MASKB);
  float*    outpre = (float*)(ws + OFF_OUTPRE);
  float*    stats  = (float*)(ws + OFF_STATS);
  _Float16* pk     = (_Float16*)(ws + OFF_PK);

  k_init_stats<<<1, 256, 0, stream>>>(stats);
  k_pack<<<(NFRAG * 32 + 63) / 64, 64, 0, stream>>>(w_off1, w_mask1, w_off2, w_mask2, pk);
  k_tox<<<NELEM / 256, 256, 0, stream>>>(x, xh);
  k_conv1<<<Bn * Hn, 256, 0, stream>>>(xh, pk, t_off, t_mask);
  k_conv3<<<Bn * Hn, 256, 0, stream>>>(t_off, t_mask, pk, offb, maskb);
  k_deform<<<NELEM / 256, 256, 0, stream>>>(x, offb, maskb, weight, bias, outpre, stats);
  k_gn_gelu<<<NELEM / 256, 256, 0, stream>>>(outpre, stats, gamma, beta, (float*)d_out);
}
